// ClusteringLayer_37005438222828
// MI455X (gfx1250) — compile-verified, matrix operand order checked by reading, not run on hardware
//
#include <hip/hip_runtime.h>
#include <stdint.h>

typedef __attribute__((ext_vector_type(16))) __bf16       v16bf;
typedef __attribute__((ext_vector_type(8)))  float        v8f;
typedef __attribute__((ext_vector_type(4)))  float        v4f;
typedef __attribute__((ext_vector_type(4)))  unsigned int v4u;
typedef __attribute__((ext_vector_type(4)))  int          v4i_t;

#define N_ROWS   65536
#define DDIM     256
#define KCLUST   256
#define WAVES    4
#define ROWS_WAVE 32   // two 16-row WMMA tiles per wave
#define ROWS_WG  128   // 4 waves * 32 rows each
#define NT       16    // 256 cluster-cols / 16 per WMMA tile
#define DT       8     // 256 contraction dim / 32 per WMMA

// ---- CDNA5 async global->LDS copy (ASYNCcnt-tracked DMA) ------------------
#if __has_builtin(__builtin_amdgcn_global_load_async_to_lds_b128)
#define GLOBAL_ASYNC_B128(g, l)                                               \
  __builtin_amdgcn_global_load_async_to_lds_b128(                             \
      (__attribute__((address_space(1))) v4i_t*)(uintptr_t)(g),               \
      (__attribute__((address_space(3))) v4i_t*)(unsigned)(uintptr_t)(l),     \
      0, 0)
#else
#define GLOBAL_ASYNC_B128(g, l)                                               \
  asm volatile("global_load_async_to_lds_b128 %0, %1, off"                    \
               :: "v"((unsigned)(uintptr_t)(l)),                              \
                  "v"((unsigned long long)(uintptr_t)(g)) : "memory")
#endif

#if __has_builtin(__builtin_amdgcn_s_wait_asynccnt)
#define WAIT_ASYNC(n) do { __builtin_amdgcn_s_wait_asynccnt(n); \
                           asm volatile("" ::: "memory"); } while (0)
#else
#define WAIT_ASYNC(n) asm volatile("s_wait_asynccnt %0" :: "n"(n) : "memory")
#endif

union Frag { v4u u4[2]; v16bf v; };

__device__ __forceinline__ v16bf load_frag(const __bf16* p) {
  // Two 16B LDS loads: elements [0..7] and [16..23] past p (K pattern of the
  // 16-bit 16x32 A/B WMMA layout: VGPR0-3 = K base..base+7, VGPR4-7 = K+16).
  Frag f;
  f.u4[0] = *(const v4u*)(p);
  f.u4[1] = *(const v4u*)(p + 16);
  return f.v;
}

// DMA one 32-row x 128B x-tile (row stride 1KB global, 128B LDS):
// 8 x async_b128, each instruction moves 32 lanes * 16B = 4 rows.
__device__ __forceinline__ void async_tile32(const float* grow, float* lrow,
                                             int lane) {
  const int r = lane >> 3;            // 0..3: row within 4-row chunk
  const int b = (lane & 7) * 4;       // float offset within 128B row
#pragma unroll
  for (int i = 0; i < 8; ++i) {
    const float* g = grow + (size_t)(i * 4 + r) * DDIM + b;
    float*       l = lrow + (i * 4 + r) * 32 + b;
    GLOBAL_ASYNC_B128(g, l);
  }
}

__global__ __launch_bounds__(128, 1)
void clustering_wmma_kernel(const float* __restrict__ x,
                            const float* __restrict__ clusters,
                            float* __restrict__ out) {
  // 289 KB static LDS (<= 320 KB per workgroup on CDNA5)
  __shared__ __align__(16) __bf16 sC_hi[KCLUST * DDIM];      // 128 KB
  __shared__ __align__(16) __bf16 sC_lo[KCLUST * DDIM];      // 128 KB
  __shared__ float                sC2[KCLUST];               //   1 KB
  __shared__ __align__(16) float  sAf[WAVES][2][ROWS_WAVE * 32]; // 32 KB dblbuf

  const int tid  = threadIdx.x;
  const int wave = tid >> 5;
  const int lane = tid & 31;
  const int l16  = lane & 15;
  const int half = lane >> 4;
  const int row0 = blockIdx.x * ROWS_WG + wave * ROWS_WAVE;

  // ---- kick off async DMA of the first x tile; overlaps cluster staging ----
  async_tile32(x + (size_t)row0 * DDIM, &sAf[wave][0][0], lane);

  // ---- Stage clusters f32 -> (bf16 hi, bf16 residual lo) in LDS, coalesced --
  for (int e = tid; e < KCLUST * DDIM; e += 128) {
    float f = clusters[e];
    __bf16 h = (__bf16)f;
    sC_hi[e] = h;
    sC_lo[e] = (__bf16)(f - (float)h);
  }
  __syncthreads();

  // ---- c2[k] = ||cluster_k||^2 (hi+lo reconstruction ~ f32 accurate) ----
  for (int k = tid; k < KCLUST; k += 128) {
    float s = 0.f;
    for (int d = 0; d < DDIM; ++d) {
      float v = (float)sC_hi[k * DDIM + d] + (float)sC_lo[k * DDIM + d];
      s += v * v;
    }
    sC2[k] = s;
  }
  __syncthreads();

  // ---- per-wave accumulators: 32 rows x 256 clusters = 2x16 WMMA C tiles ----
  v8f acc[2][NT];
  const v8f vzero = {0.f, 0.f, 0.f, 0.f, 0.f, 0.f, 0.f, 0.f};
#pragma unroll
  for (int m = 0; m < 2; ++m)
#pragma unroll
    for (int nt = 0; nt < NT; ++nt) acc[m][nt] = vzero;

  float x2p[2] = {0.f, 0.f};  // partial ||x_row||^2 (rows l16 / 16+l16)

#pragma unroll 1
  for (int dt = 0; dt < DT; ++dt) {
    const int d0  = dt * 32;
    const int buf = dt & 1;

    // prefetch next tile into the other buffer, then gate on current tile
    if (dt + 1 < DT) {
      async_tile32(x + (size_t)row0 * DDIM + (d0 + 32), &sAf[wave][buf ^ 1][0],
                   lane);
      WAIT_ASYNC(8);   // the 8 just-issued may remain in flight
    } else {
      WAIT_ASYNC(0);
    }

    // A fragments for both 16-row tiles:
    // lane l -> row (m*16 + (l&15)); K = (l>>4)*8..+7 and +16..
    v16bf a_hi[2], a_lo[2];
#pragma unroll
    for (int m = 0; m < 2; ++m) {
      const float* abase =
          &sAf[wave][buf][(m * 16 + l16) * 32 + half * 8];
      v4f q0 = *(const v4f*)(abase + 0);
      v4f q1 = *(const v4f*)(abase + 4);
      v4f q2 = *(const v4f*)(abase + 16);
      v4f q3 = *(const v4f*)(abase + 20);
      float fa[16];
#pragma unroll
      for (int e = 0; e < 4; ++e) {
        fa[e]      = q0[e];
        fa[4 + e]  = q1[e];
        fa[8 + e]  = q2[e];
        fa[12 + e] = q3[e];
      }
#pragma unroll
      for (int e = 0; e < 16; ++e) {
        float f = fa[e];
        __bf16 h = (__bf16)f;
        a_hi[m][e] = h;
        a_lo[m][e] = (__bf16)(f - (float)h);
        x2p[m] += f * f;            // exact f32 row-norm accumulation
      }
    }

#pragma unroll
    for (int nt = 0; nt < NT; ++nt) {
      // B = clusters^T : B[k][n] = clusters[n][k]; mirrored per-lane K pattern
      const __bf16* pbh = &sC_hi[(nt * 16 + l16) * DDIM + d0 + half * 8];
      const __bf16* pbl = &sC_lo[(nt * 16 + l16) * DDIM + d0 + half * 8];
      v16bf b_hi = load_frag(pbh);
      v16bf b_lo = load_frag(pbl);

      // each B hi/lo pair feeds 6 WMMAs (both row tiles): 3-term bf16 split
#pragma unroll
      for (int m = 0; m < 2; ++m) {
        acc[m][nt] = __builtin_amdgcn_wmma_f32_16x16x32_bf16(
            false, a_hi[m], false, b_hi, (short)0, acc[m][nt], false, false);
        acc[m][nt] = __builtin_amdgcn_wmma_f32_16x16x32_bf16(
            false, a_hi[m], false, b_lo, (short)0, acc[m][nt], false, false);
        acc[m][nt] = __builtin_amdgcn_wmma_f32_16x16x32_bf16(
            false, a_lo[m], false, b_hi, (short)0, acc[m][nt], false, false);
      }
    }
  }

  // ---- finish ||x_row||^2, q = 1/(1+dist2), row-normalize, store ----
#pragma unroll
  for (int m = 0; m < 2; ++m) {
    // lane r and lane r+16 both end up holding row (m*16+r)'s full sum
    float s2 = x2p[m] + __shfl_xor(x2p[m], 16, 32);
    float x2r[8];
#pragma unroll
    for (int j = 0; j < 8; ++j)
      x2r[j] = __shfl(s2, j + 8 * half, 32);

    float rs[8];
#pragma unroll
    for (int j = 0; j < 8; ++j) rs[j] = 0.f;

#pragma unroll
    for (int nt = 0; nt < NT; ++nt) {
      float c2v = sC2[nt * 16 + l16];
#pragma unroll
      for (int j = 0; j < 8; ++j) {
        // C/D layout: VGPR j, lanes 0-15 -> M=j ; lanes 16-31 -> M=j+8
        float dist2 = x2r[j] + c2v - 2.f * acc[m][nt][j];
        float qv = 1.f / (1.f + dist2);
        acc[m][nt][j] = qv;
        rs[j] += qv;
      }
    }
    // sum across the 16 lanes of each half-wave (xor 1,2,4,8 stays in half)
#pragma unroll
    for (int j = 0; j < 8; ++j) {
      float s = rs[j];
      s += __shfl_xor(s, 1, 32);
      s += __shfl_xor(s, 2, 32);
      s += __shfl_xor(s, 4, 32);
      s += __shfl_xor(s, 8, 32);
      rs[j] = 1.f / s;
    }

#pragma unroll
    for (int nt = 0; nt < NT; ++nt) {
#pragma unroll
      for (int j = 0; j < 8; ++j) {
        int row = row0 + m * 16 + j + 8 * half;
        int col = nt * 16 + l16;
        out[(size_t)row * KCLUST + col] = acc[m][nt][j] * rs[j];
      }
    }
  }
}

extern "C" void kernel_launch(void* const* d_in, const int* in_sizes, int n_in,
                              void* d_out, int out_size, void* d_ws, size_t ws_size,
                              hipStream_t stream) {
  const float* x        = (const float*)d_in[0];
  const float* clusters = (const float*)d_in[1];
  float* out            = (float*)d_out;
  (void)in_sizes; (void)n_in; (void)out_size; (void)d_ws; (void)ws_size;

  dim3 grid(N_ROWS / ROWS_WG);  // 512 workgroups
  dim3 block(128);              // 4 wave32: 1 wave/SIMD, full VGPR budget
  hipLaunchKernelGGL(clustering_wmma_kernel, grid, block, 0, stream,
                     x, clusters, out);
}
